// Decoder_17154099381038
// MI455X (gfx1250) — matrix operand order repaired and checked by reading, not compiled, not run on hardware
//
#include <hip/hip_runtime.h>

// ---------------------------------------------------------------------------
// 2-layer transformer decoder forward (B=1, S=2048, D=512, H=8, dh=64, F=2048)
// CDNA5 / gfx1250, wave32. All GEMMs + attention via v_wmma_f32_16x16x32_f16,
// f16 operands staged with b128 loads, f32 accumulation, flash attention.
// ---------------------------------------------------------------------------

#define D_  512
#define S_  2048
#define H_  8
#define DH_ 64
#define F_  2048
#define NEGV (-1e9f)

typedef __attribute__((ext_vector_type(16))) _Float16 v16h;
typedef __attribute__((ext_vector_type(8)))  float    v8f;

union AFrag { v16h v; unsigned int u[8]; uint4 q[2]; };

__device__ __forceinline__ v8f wmma_f16(v16h a, v16h b, v8f c) {
  // 8 args: (neg_a, A, neg_b, B, c_mod, C, reuse_a, reuse_b)
  return __builtin_amdgcn_wmma_f32_16x16x32_f16(false, a, false, b, (short)0, c,
                                                false, false);
}

// A-fragment (16x32 f16): lane m=l%16, half=l/16.
//   u[0..3] = 8 contiguous halves at k = 16*half
//   u[4..7] = 8 contiguous halves at k = 16*half + 8
// B-fragment (32x16 f16, gathered from K-major storage): lane n=l%16.
//   u[0..7] = 16 contiguous halves at k = 16*half

// ---------------------------------------------------------------------------
// Weight prep: dst[N][K] f16 = transpose(src[K][N] f32). 32x32 tiles via LDS.
// ---------------------------------------------------------------------------
__global__ __launch_bounds__(256) void wt_prep(const float* __restrict__ src,
                                               _Float16* __restrict__ dst,
                                               int K, int N) {
  __shared__ float tile[32][33];
  const int tx = threadIdx.x & 31;
  const int ty = threadIdx.x >> 5;  // 0..7
  const int n = blockIdx.x * 32 + tx;
  const int kb = blockIdx.y * 32;
#pragma unroll
  for (int i = ty; i < 32; i += 8) tile[i][tx] = src[(kb + i) * N + n];
  __syncthreads();
  const int k = kb + tx;
  const int nb = blockIdx.x * 32;
#pragma unroll
  for (int i = ty; i < 32; i += 8) dst[(nb + i) * K + k] = (_Float16)tile[tx][i];
}

// ---------------------------------------------------------------------------
// f32 -> f16 elementwise
// ---------------------------------------------------------------------------
__global__ __launch_bounds__(256) void conv16(const float* __restrict__ in,
                                              _Float16* __restrict__ out, int n) {
  int i = blockIdx.x * 256 + threadIdx.x;
  if (i < n) out[i] = (_Float16)in[i];
}

// ---------------------------------------------------------------------------
// Positional encoding: out = x + PE ([sin(ang even cols) | cos(ang odd cols)])
// ---------------------------------------------------------------------------
__global__ __launch_bounds__(256) void posenc_kernel(const float* __restrict__ x,
                                                     float* __restrict__ out,
                                                     _Float16* __restrict__ out16) {
  int idx = blockIdx.x * 256 + threadIdx.x;  // over S*D
  int srow = idx >> 9;
  int c = idx & 511;
  float pos = (float)srow;
  float pe;
  if (c < 256) {
    float rate = powf(10000.0f, -(2.0f * (float)c) / 512.0f);
    pe = sinf(pos * rate);
  } else {
    int j = c - 256;
    float rate = powf(10000.0f, -(2.0f * (float)j) / 512.0f);
    pe = cosf(pos * rate);
  }
  float v = x[idx] + pe;
  out[idx] = v;
  out16[idx] = (_Float16)v;
}

// ---------------------------------------------------------------------------
// WMMA GEMM: C[M,N] = A[M,K](f16) @ WT[N,K](f16, pre-transposed) + bias.
// 256 threads = 8 waves. Tile BM=128 x BN=64, BK=32. Wave w: rows 16w..16w+15,
// all 4 n-tiles. Register-prefetch software pipeline over k; all fragments
// loaded before the 4-WMMA burst (single dscnt wait per k-step).
// ---------------------------------------------------------------------------
__global__ __launch_bounds__(256) void gemm_f16(
    const _Float16* __restrict__ A, const _Float16* __restrict__ WT,
    const float* __restrict__ bias, float* __restrict__ C32,
    _Float16* __restrict__ C16, int M, int N, int K, int relu) {
  __shared__ __align__(16) _Float16 As[128 * 40];  // [m][k], stride 40 halves
  __shared__ __align__(16) _Float16 Bs[64 * 40];   // [n][k], stride 40 halves

  const int t = threadIdx.x;
  const int lane = t & 31;
  const int wave = t >> 5;
  const int ml = lane & 15;
  const int half = lane >> 4;
  const int m0 = blockIdx.y * 128;
  const int n0 = blockIdx.x * 64;

  // staging: A tile 128x32 halves = 2 uint4/thread, B tile 64x32 = 1 uint4/thread
  const int ar0 = t >> 2;          // 0..63
  const int ar1 = 64 + (t >> 2);   // 64..127
  const int kc = (t & 3) * 8;      // 0,8,16,24
  const _Float16* pa0 = A + (m0 + ar0) * K + kc;
  const _Float16* pa1 = A + (m0 + ar1) * K + kc;
  const _Float16* pb = WT + (n0 + ar0) * K + kc;

  uint4 ra0 = *(const uint4*)pa0;
  uint4 ra1 = *(const uint4*)pa1;
  uint4 rb = *(const uint4*)pb;

  v8f acc[4] = {};
  const int arow = 16 * wave + ml;

  for (int k0 = 0; k0 < K; k0 += 32) {
    *(uint4*)&As[ar0 * 40 + kc] = ra0;
    *(uint4*)&As[ar1 * 40 + kc] = ra1;
    *(uint4*)&Bs[ar0 * 40 + kc] = rb;
    __syncthreads();
    if (k0 + 32 < K) {  // prefetch next k-tile while WMMAs run
      pa0 += 32; pa1 += 32; pb += 32;
      ra0 = *(const uint4*)pa0;
      ra1 = *(const uint4*)pa1;
      rb = *(const uint4*)pb;
    }
    // load ALL fragments first, then issue WMMAs back-to-back
    AFrag af, bf[4];
    af.q[0] = *(const uint4*)&As[arow * 40 + 16 * half];
    af.q[1] = *(const uint4*)&As[arow * 40 + 16 * half + 8];
#pragma unroll
    for (int nt = 0; nt < 4; ++nt) {
      bf[nt].q[0] = *(const uint4*)&Bs[(16 * nt + ml) * 40 + 16 * half];
      bf[nt].q[1] = *(const uint4*)&Bs[(16 * nt + ml) * 40 + 16 * half + 8];
    }
#pragma unroll
    for (int nt = 0; nt < 4; ++nt) acc[nt] = wmma_f16(af.v, bf[nt].v, acc[nt]);
    __syncthreads();
  }

#pragma unroll
  for (int nt = 0; nt < 4; ++nt) {
    int ng = n0 + 16 * nt + ml;
    float bv = bias ? bias[ng] : 0.0f;
#pragma unroll
    for (int r = 0; r < 8; ++r) {
      int rg = m0 + 16 * wave + r + 8 * half;
      float v = acc[nt][r] + bv;
      if (relu) v = fmaxf(v, 0.0f);
      if (C32) C32[rg * N + ng] = v;
      if (C16) C16[rg * N + ng] = (_Float16)v;
    }
  }
}

// ---------------------------------------------------------------------------
// f16 transpose: out[d*S + s] = in[s*D + d]   (V -> V^T for PV fragments)
// ---------------------------------------------------------------------------
__global__ __launch_bounds__(256) void transpose_h(const _Float16* __restrict__ in,
                                                   _Float16* __restrict__ out) {
  int idx = blockIdx.x * 256 + threadIdx.x;  // over S*D
  int s = idx >> 9;
  int d = idx & 511;
  out[d * S_ + s] = in[idx];
}

// ---------------------------------------------------------------------------
// Flash attention: one wave per (16-query tile, head). Q,K f16 [S][D],
// VT f16 [D][S]. 32-key blocks, online softmax, f16 output [S][D].
// Fragment loads batched ahead of each 4-WMMA burst.
// ---------------------------------------------------------------------------
__global__ __launch_bounds__(32) void attn_flash(
    const _Float16* __restrict__ Q, const _Float16* __restrict__ Kg,
    const _Float16* __restrict__ VT, _Float16* __restrict__ O16, int Skv,
    int causal) {
  __shared__ __align__(16) _Float16 Pl[16 * 40];  // P-tile layout bounce

  const int lane = threadIdx.x & 31;
  const int ml = lane & 15;
  const int half = lane >> 4;
  const int q0 = blockIdx.x * 16;
  const int h = blockIdx.y;

  // Q fragments: two 16x32 A-fragments, b128 global gathers
  AFrag aq[2];
#pragma unroll
  for (int tt = 0; tt < 2; ++tt) {
    const _Float16* qp = &Q[(q0 + ml) * D_ + h * DH_ + 32 * tt + 16 * half];
    aq[tt].q[0] = *(const uint4*)qp;
    aq[tt].q[1] = *(const uint4*)(qp + 8);
  }

  v8f o[4] = {};
  float mrow[8], lrow[8];
#pragma unroll
  for (int r = 0; r < 8; ++r) { mrow[r] = -3.0e38f; lrow[r] = 0.0f; }

  const int kend = causal ? (q0 + 16) : Skv;
  for (int kb = 0; kb < kend; kb += 32) {
    // ---- logits: S = Q @ K^T over 32 keys. Load 4 K fragments, then 4 WMMAs.
    AFrag bk[4];  // [jn][tt] flattened: jn*2+tt
#pragma unroll
    for (int jn = 0; jn < 2; ++jn)
#pragma unroll
      for (int tt = 0; tt < 2; ++tt) {
        const _Float16* kp =
            &Kg[(kb + 16 * jn + ml) * D_ + h * DH_ + 32 * tt + 16 * half];
        bk[jn * 2 + tt].q[0] = *(const uint4*)kp;
        bk[jn * 2 + tt].q[1] = *(const uint4*)(kp + 8);
      }
    v8f sA[2] = {};
#pragma unroll
    for (int jn = 0; jn < 2; ++jn)
#pragma unroll
      for (int tt = 0; tt < 2; ++tt)
        sA[jn] = wmma_f16(aq[tt].v, bk[jn * 2 + tt].v, sA[jn]);

    // ---- scale + causal mask + online softmax ------------------------------
    float p0[8], p1[8], mx[8], alp[8], rs[8];
#pragma unroll
    for (int r = 0; r < 8; ++r) {
      float v0 = sA[0][r] * 0.125f;  // 1/sqrt(64)
      float v1 = sA[1][r] * 0.125f;
      if (causal) {
        int qrow = q0 + r + 8 * half;
        if (kb + ml > qrow)      v0 += NEGV;
        if (kb + 16 + ml > qrow) v1 += NEGV;
      }
      p0[r] = v0; p1[r] = v1;
      mx[r] = fmaxf(v0, v1);
    }
#pragma unroll
    for (int sh = 1; sh < 16; sh <<= 1)
#pragma unroll
      for (int r = 0; r < 8; ++r) mx[r] = fmaxf(mx[r], __shfl_xor(mx[r], sh, 32));

#pragma unroll
    for (int r = 0; r < 8; ++r) {
      float mn = fmaxf(mrow[r], mx[r]);
      alp[r] = __expf(mrow[r] - mn);
      mrow[r] = mn;
      p0[r] = __expf(p0[r] - mn);
      p1[r] = __expf(p1[r] - mn);
      rs[r] = p0[r] + p1[r];
    }
#pragma unroll
    for (int sh = 1; sh < 16; sh <<= 1)
#pragma unroll
      for (int r = 0; r < 8; ++r) rs[r] += __shfl_xor(rs[r], sh, 32);
#pragma unroll
    for (int r = 0; r < 8; ++r) lrow[r] = lrow[r] * alp[r] + rs[r];
#pragma unroll
    for (int nt = 0; nt < 4; ++nt)
#pragma unroll
      for (int r = 0; r < 8; ++r) o[nt][r] *= alp[r];

    // ---- convert P (C-layout) -> A-fragment through LDS --------------------
#pragma unroll
    for (int r = 0; r < 8; ++r) {
      Pl[(r + 8 * half) * 40 + ml] = (_Float16)p0[r];
      Pl[(r + 8 * half) * 40 + 16 + ml] = (_Float16)p1[r];
    }
    asm volatile("s_wait_dscnt 0" ::: "memory");  // same-wave DS RAW ordering
    AFrag pa;
    pa.q[0] = *(const uint4*)&Pl[ml * 40 + 16 * half];
    pa.q[1] = *(const uint4*)&Pl[ml * 40 + 16 * half + 8];

    // ---- O += P @ V: load 4 V fragments, then 4 WMMAs ----------------------
    AFrag bv[4];
#pragma unroll
    for (int nt = 0; nt < 4; ++nt) {
      const _Float16* vp = &VT[(h * DH_ + 16 * nt + ml) * Skv + kb + 16 * half];
      bv[nt].q[0] = *(const uint4*)vp;
      bv[nt].q[1] = *(const uint4*)(vp + 8);
    }
#pragma unroll
    for (int nt = 0; nt < 4; ++nt) o[nt] = wmma_f16(pa.v, bv[nt].v, o[nt]);
  }

  float rinv[8];
#pragma unroll
  for (int r = 0; r < 8; ++r) rinv[r] = 1.0f / lrow[r];
#pragma unroll
  for (int nt = 0; nt < 4; ++nt)
#pragma unroll
    for (int r = 0; r < 8; ++r)
      O16[(q0 + r + 8 * half) * D_ + h * DH_ + 16 * nt + ml] =
          (_Float16)(o[nt][r] * rinv[r]);
}

// ---------------------------------------------------------------------------
// Fused residual + LayerNorm: out = LN(X + R)*g + b  (f32 + optional f16 out)
// ---------------------------------------------------------------------------
__global__ __launch_bounds__(128) void ln_fused(
    const float* __restrict__ X, const float* __restrict__ R,
    const float* __restrict__ g, const float* __restrict__ b,
    float* __restrict__ out, _Float16* __restrict__ out16) {
  __shared__ float red[4];
  const int row = blockIdx.x, t = threadIdx.x;
  const int lane = t & 31, w = t >> 5;
  float v[4];
  float s = 0.0f;
#pragma unroll
  for (int i = 0; i < 4; ++i) {
    int c = t + 128 * i;
    v[i] = X[row * D_ + c] + R[row * D_ + c];
    s += v[i];
  }
#pragma unroll
  for (int sh = 1; sh < 32; sh <<= 1) s += __shfl_xor(s, sh, 32);
  if (lane == 0) red[w] = s;
  __syncthreads();
  float mu = (red[0] + red[1] + red[2] + red[3]) * (1.0f / D_);
  __syncthreads();
  float vs = 0.0f;
#pragma unroll
  for (int i = 0; i < 4; ++i) { float dd = v[i] - mu; vs += dd * dd; }
#pragma unroll
  for (int sh = 1; sh < 32; sh <<= 1) vs += __shfl_xor(vs, sh, 32);
  if (lane == 0) red[w] = vs;
  __syncthreads();
  float var = (red[0] + red[1] + red[2] + red[3]) * (1.0f / D_);
  float inv = rsqrtf(var + 1e-6f);
#pragma unroll
  for (int i = 0; i < 4; ++i) {
    int c = t + 128 * i;
    float o = (v[i] - mu) * inv * g[c] + b[c];
    out[row * D_ + c] = o;
    if (out16) out16[row * D_ + c] = (_Float16)o;
  }
}

// ---------------------------------------------------------------------------
// Host orchestration
// ---------------------------------------------------------------------------
static inline void launch_gemm(const _Float16* A, const _Float16* WT,
                               const float* bias, float* C32, _Float16* C16,
                               int M, int N, int K, int relu, hipStream_t s) {
  dim3 grid(N / 64, M / 128), blk(256);
  gemm_f16<<<grid, blk, 0, s>>>(A, WT, bias, C32, C16, M, N, K, relu);
}

extern "C" void kernel_launch(void* const* d_in, const int* in_sizes, int n_in,
                              void* d_out, int out_size, void* d_ws, size_t ws_size,
                              hipStream_t stream) {
  (void)in_sizes; (void)n_in; (void)out_size; (void)ws_size;
  const float* x_in = (const float*)d_in[0];
  const float* enc  = (const float*)d_in[1];
  const float* a1w  = (const float*)d_in[2];
  const float* a1b  = (const float*)d_in[3];
  const float* a2w  = (const float*)d_in[4];
  const float* a2b  = (const float*)d_in[5];
  const float* fw1  = (const float*)d_in[6];
  const float* fb1  = (const float*)d_in[7];
  const float* fw2  = (const float*)d_in[8];
  const float* fb2  = (const float*)d_in[9];
  const float* lng  = (const float*)d_in[10];
  const float* lnb  = (const float*)d_in[11];

  char* wsp = (char*)d_ws;
  auto alloc = [&](size_t bytes) {
    char* p = wsp;
    wsp += (bytes + 255) & ~(size_t)255;
    return p;
  };
  float*    bufAf = (float*)   alloc((size_t)S_ * D_ * 4);
  float*    bufBf = (float*)   alloc((size_t)S_ * D_ * 4);
  _Float16* bufAh = (_Float16*)alloc((size_t)S_ * D_ * 2);
  _Float16* bufBh = (_Float16*)alloc((size_t)S_ * D_ * 2);
  float*    tmp   = (float*)   alloc((size_t)S_ * D_ * 4);
  _Float16* att16 = (_Float16*)alloc((size_t)S_ * D_ * 2);
  _Float16* q16   = (_Float16*)alloc((size_t)S_ * D_ * 2);
  _Float16* k16   = (_Float16*)alloc((size_t)S_ * D_ * 2);
  _Float16* v16   = (_Float16*)alloc((size_t)S_ * D_ * 2);
  _Float16* vT16  = (_Float16*)alloc((size_t)S_ * D_ * 2);
  _Float16* enc16 = (_Float16*)alloc((size_t)S_ * D_ * 2);
  _Float16* hid16 = (_Float16*)alloc((size_t)S_ * F_ * 2);
  _Float16* wtbuf = (_Float16*)alloc((size_t)2 * (8 * D_ * D_ + 2 * D_ * F_) * 2);

  // ---- weight prep: f32 [K][N] -> f16 [N][K] for every matrix --------------
  _Float16* wptr = wtbuf;
  auto prep = [&](const float* src, int K, int N) {
    _Float16* dst = wptr;
    wptr += (size_t)K * N;
    dim3 g(N / 32, K / 32);
    wt_prep<<<g, 256, 0, stream>>>(src, dst, K, N);
    return dst;
  };
  _Float16 *wt1[2][4], *wt2[2][4], *wtf1[2], *wtf2[2];
  const size_t DD = (size_t)D_ * D_;
  for (int l = 0; l < 2; ++l) {
    for (int i = 0; i < 4; ++i) wt1[l][i] = prep(a1w + ((size_t)l * 4 + i) * DD, D_, D_);
    for (int i = 0; i < 4; ++i) wt2[l][i] = prep(a2w + ((size_t)l * 4 + i) * DD, D_, D_);
    wtf1[l] = prep(fw1 + (size_t)l * D_ * F_, D_, F_);  // -> [F][D]
    wtf2[l] = prep(fw2 + (size_t)l * F_ * D_, F_, D_);  // -> [D][F]
  }
  conv16<<<(S_ * D_) / 256, 256, 0, stream>>>(enc, enc16, S_ * D_);
  posenc_kernel<<<(S_ * D_) / 256, 256, 0, stream>>>(x_in, bufAf, bufAh);

  float*    curF = bufAf; _Float16* curH = bufAh;
  float*    altF = bufBf; _Float16* altH = bufBh;
  const dim3 agrid(S_ / 16, H_);
  for (int l = 0; l < 2; ++l) {
    const float* b1p = a1b + (size_t)(l * 4) * D_;
    const float* b2p = a2b + (size_t)(l * 4) * D_;
    // ---- self attention ----
    launch_gemm(curH, wt1[l][0], b1p + 0 * D_, nullptr, q16, S_, D_, D_, 0, stream);
    launch_gemm(curH, wt1[l][1], b1p + 1 * D_, nullptr, k16, S_, D_, D_, 0, stream);
    launch_gemm(curH, wt1[l][2], b1p + 2 * D_, nullptr, v16, S_, D_, D_, 0, stream);
    transpose_h<<<(S_ * D_) / 256, 256, 0, stream>>>(v16, vT16);
    attn_flash<<<agrid, 32, 0, stream>>>(q16, k16, vT16, att16, S_, 1);
    launch_gemm(att16, wt1[l][3], b1p + 3 * D_, tmp, nullptr, S_, D_, D_, 0, stream);
    ln_fused<<<S_, 128, 0, stream>>>(curF, tmp, lng + (l * 3 + 0) * D_,
                                     lnb + (l * 3 + 0) * D_, altF, altH);
    { float* tf = curF; curF = altF; altF = tf; }
    { _Float16* th = curH; curH = altH; altH = th; }

    // ---- cross attention ----
    launch_gemm(curH, wt2[l][0], b2p + 0 * D_, nullptr, q16, S_, D_, D_, 0, stream);
    launch_gemm(enc16, wt2[l][1], b2p + 1 * D_, nullptr, k16, S_, D_, D_, 0, stream);
    launch_gemm(enc16, wt2[l][2], b2p + 2 * D_, nullptr, v16, S_, D_, D_, 0, stream);
    transpose_h<<<(S_ * D_) / 256, 256, 0, stream>>>(v16, vT16);
    attn_flash<<<agrid, 32, 0, stream>>>(q16, k16, vT16, att16, S_, 0);
    launch_gemm(att16, wt2[l][3], b2p + 3 * D_, tmp, nullptr, S_, D_, D_, 0, stream);
    ln_fused<<<S_, 128, 0, stream>>>(curF, tmp, lng + (l * 3 + 1) * D_,
                                     lnb + (l * 3 + 1) * D_, altF, altH);
    { float* tf = curF; curF = altF; altF = tf; }
    { _Float16* th = curH; curH = altH; altH = th; }

    // ---- FFN ----
    launch_gemm(curH, wtf1[l], fb1 + (size_t)l * F_, nullptr, hid16, S_, F_, D_,
                1, stream);
    launch_gemm(hid16, wtf2[l], fb2 + (size_t)l * D_, tmp, nullptr, S_, D_, F_,
                0, stream);
    float* out3F = (l == 1) ? (float*)d_out : altF;
    _Float16* out3H = (l == 1) ? nullptr : altH;
    ln_fused<<<S_, 128, 0, stream>>>(curF, tmp, lng + (l * 3 + 2) * D_,
                                     lnb + (l * 3 + 2) * D_, out3F, out3H);
    { float* tf = curF; curF = out3F; altF = tf; }
    { _Float16* th = curH; curH = out3H; altH = th; }
  }
}
